// PyGNN_68796786147750
// MI455X (gfx1250) — matrix operand order repaired and checked by reading, not compiled, not run on hardware
//
#include <hip/hip_runtime.h>
#include <hip/hip_bf16.h>
#include <math.h>

typedef __attribute__((ext_vector_type(16))) _Float16 v16h;
typedef __attribute__((ext_vector_type(8)))  _Float16 v8h;
typedef __attribute__((ext_vector_type(8)))  float    v8f;

#define NN 50000
#define FF 128
#define HH 64
#define BB 4
#define EE 500000
#define FH 192   // F + H
#define BH 256   // B * H

// ---------------- utility fill ----------------
__global__ void fill_f32(float* __restrict__ p, float v, int n) {
    int i = blockIdx.x * blockDim.x + threadIdx.x;
    if (i < n) p[i] = v;
}

// ---------------- time encode + scatter-add at targets ----------------
// t_agg[col[e]][h] += cos(edge_attr[e] * te_w[h] + te_b[h])
__global__ void time_encode_scatter(const float* __restrict__ edge_attr,
                                    const int* __restrict__ edge_index,
                                    const float* __restrict__ te_w,
                                    const float* __restrict__ te_b,
                                    float* __restrict__ t_agg) {
    int idx = blockIdx.x * blockDim.x + threadIdx.x;   // E*64 threads
    if (idx >= EE * 64) return;
    int e = idx >> 6;
    int h = idx & 63;
    float t = edge_attr[e];
    float val = cosf(t * te_w[h] + te_b[h]);
    int col = edge_index[EE + e];                       // edge_index[1]
    atomicAdd(&t_agg[col * HH + h], val);
}

// ---------------- per-band target-degree (deg pre-initialized to 1 = self loop) --
__global__ void band_degree(const int* __restrict__ sub_edges, float* __restrict__ deg) {
    int idx = blockIdx.x * blockDim.x + threadIdx.x;    // B*E threads
    if (idx >= BB * EE) return;
    int b = idx / EE;
    int e = idx - b * EE;
    int col = sub_edges[b * 2 * EE + EE + e];
    atomicAdd(&deg[b * NN + col], 1.0f);
}

__global__ void deg_to_dinv(float* __restrict__ deg) {
    int i = blockIdx.x * blockDim.x + threadIdx.x;      // B*N threads
    if (i < BB * NN) deg[i] = rsqrtf(deg[i]);           // deg >= 1 always
}

// ---------------- pre-swizzle weights into WMMA B-fragment order ----------------
// dst[((ntile*KS + ks)*32 + lane)*16 + i] = W[k][n]  (f16)
// with n = ntile*16 + lane%16 ; k = ks*32 + (lane>=16 ? 8 : 0) + (i<8 ? i : i+8)
__global__ void pack_taw(const float* __restrict__ ta_w, _Float16* __restrict__ dst) {
    int idx = blockIdx.x * blockDim.x + threadIdx.x;    // 4*2*32*16 = 4096
    if (idx >= 4096) return;
    int i    = idx & 15;
    int lane = (idx >> 4) & 31;
    int ks   = (idx >> 9) & 1;
    int nt   = idx >> 10;
    int n = nt * 16 + (lane & 15);
    int k = ks * 32 + ((lane & 16) ? 8 : 0) + (i < 8 ? i : i + 8);
    dst[idx] = (_Float16)ta_w[k * HH + n];
}

__global__ void pack_convw(const float* __restrict__ conv_w, _Float16* __restrict__ dst) {
    int idx = blockIdx.x * blockDim.x + threadIdx.x;    // 16*6*32*16 = 49152
    if (idx >= 49152) return;
    int i    = idx & 15;
    int lane = (idx >> 4) & 31;
    int ks   = (idx >> 9) % 6;
    int nt   = idx / 3072;
    int ncol = nt * 16 + (lane & 15);                   // 0..255 (band*64 + j)
    int band = ncol >> 6;
    int j    = ncol & 63;
    int k    = ks * 32 + ((lane & 16) ? 8 : 0) + (i < 8 ? i : i + 8);    // 0..191
    dst[idx] = (_Float16)conv_w[band * FH * HH + k * HH + j];
}

// ---------------- feature f32 -> f16 into featH[:, 0:128] ----------------
__global__ void feat_to_half(const float* __restrict__ feature, _Float16* __restrict__ featH) {
    int idx = blockIdx.x * blockDim.x + threadIdx.x;    // N*128
    if (idx >= NN * FF) return;
    int n = idx / FF, f = idx - n * FF;
    featH[n * FH + f] = (_Float16)feature[idx];
}

// ---------------- agg GEMM: featH[:,128:192] = (t_agg @ ta_w + ta_b) (f16) ------
// one wave per 16-row stripe, all 4 n-tiles held in registers (A reused 4x).
// K = 64 -> 2 k-steps x 4 n-tiles = 8 WMMAs per wave.
__global__ void agg_gemm(const float* __restrict__ t_agg,
                         const _Float16* __restrict__ bpack,
                         const float* __restrict__ ta_b,
                         _Float16* __restrict__ featH) {
    const int mt = blockIdx.x;            // 0..3124
    const int lane = threadIdx.x;         // 0..31 (wave32)
    const int mrow = mt * 16 + (lane & 15);
    const int hi = lane >> 4;
    v8f c[4] = {};
    #pragma unroll
    for (int ks = 0; ks < 2; ++ks) {
        const float* arow = t_agg + mrow * HH + ks * 32;
        const int o0 = hi ? 8 : 0;
        v16h a;
        #pragma unroll
        for (int i = 0; i < 8; ++i) a[i] = (_Float16)arow[o0 + i];
        #pragma unroll
        for (int i = 0; i < 8; ++i) a[8 + i] = (_Float16)arow[o0 + 16 + i];
        #pragma unroll
        for (int nt = 0; nt < 4; ++nt) {
            const _Float16* bp = bpack + (((nt * 2 + ks) * 32 + lane) << 4);
            v8h b0 = *(const v8h*)bp;
            v8h b1 = *(const v8h*)(bp + 8);
            v16h b = __builtin_shufflevector(b0, b1, 0,1,2,3,4,5,6,7,8,9,10,11,12,13,14,15);
            c[nt] = __builtin_amdgcn_wmma_f32_16x16x32_f16(false, a, false, b, (short)0, c[nt], false, false);
        }
    }
    #pragma unroll
    for (int nt = 0; nt < 4; ++nt) {
        const int ncol = nt * 16 + (lane & 15);
        const float bias = ta_b[ncol];
        #pragma unroll
        for (int v = 0; v < 8; ++v) {
            int row = mt * 16 + v + (hi ? 8 : 0);
            featH[row * FH + FF + ncol] = (_Float16)(c[nt][v] + bias);
        }
    }
}

// ---------------- fused 4-band GEMM: xw[N,256] = featH[N,192] @ Wcat[192,256] ----
// 4 waves per block; each wave owns one 16x64 output slab (4 n-tiles in
// registers), so each A fragment feeds 4 WMMAs. K = 192 -> 6 k-steps,
// 24 WMMAs per wave.
__global__ void band_gemm(const _Float16* __restrict__ featH,
                          const _Float16* __restrict__ wpack,
                          float* __restrict__ xw) {
    const int mt = blockIdx.x;            // 0..3124
    const int wave = threadIdx.x >> 5;    // 0..3 -> n-range wave*64..+63
    const int lane = threadIdx.x & 31;
    const int mrow = mt * 16 + (lane & 15);
    const int hi = lane >> 4;
    const _Float16* arow = featH + mrow * FH;
    v8f c[4] = {};
    #pragma unroll
    for (int ks = 0; ks < 6; ++ks) {
        const int kb = ks * 32 + (hi ? 8 : 0);
        v8h a0 = *(const v8h*)(arow + kb);
        v8h a1 = *(const v8h*)(arow + kb + 16);
        v16h a = __builtin_shufflevector(a0, a1, 0,1,2,3,4,5,6,7,8,9,10,11,12,13,14,15);
        #pragma unroll
        for (int t = 0; t < 4; ++t) {
            const int nt = wave * 4 + t;
            const _Float16* bp = wpack + (((nt * 6 + ks) * 32 + lane) << 4);
            v8h b0 = *(const v8h*)bp;
            v8h b1 = *(const v8h*)(bp + 8);
            v16h b = __builtin_shufflevector(b0, b1, 0,1,2,3,4,5,6,7,8,9,10,11,12,13,14,15);
            c[t] = __builtin_amdgcn_wmma_f32_16x16x32_f16(false, a, false, b, (short)0, c[t], false, false);
        }
    }
    #pragma unroll
    for (int t = 0; t < 4; ++t) {
        const int ncol = (wave * 4 + t) * 16 + (lane & 15);
        #pragma unroll
        for (int v = 0; v < 8; ++v) {
            int row = mt * 16 + v + (hi ? 8 : 0);
            xw[row * BH + ncol] = c[t][v];
        }
    }
}

// ---------------- per-band normalized scatter-add (edges only) -------------------
// bandout[col][b*64+j] += xw[row][b*64+j] * dinv[b][row] * dinv[b][col]
__global__ void band_scatter(const int* __restrict__ sub_edges,
                             const float* __restrict__ dinv,
                             const float* __restrict__ xw,
                             float* __restrict__ bandout) {
    int idx = blockIdx.x * blockDim.x + threadIdx.x;    // B*E*64 = 128M threads
    int j = idx & 63;
    int t = idx >> 6;                                   // < B*E
    int b = t / EE;
    int e = t - b * EE;
    int row = sub_edges[b * 2 * EE + e];
    int col = sub_edges[b * 2 * EE + EE + e];
    float norm = dinv[b * NN + row] * dinv[b * NN + col];
    float val = xw[row * BH + b * HH + j] * norm;
    atomicAdd(&bandout[col * BH + b * HH + j], val);
}

// ---------------- finalize: self-loops + bias + relu + lin_out + log_softmax -----
// one wave per node; lanes cover the 256-wide hidden vector
__global__ void final_out(const float* __restrict__ bandout,
                          const float* __restrict__ xw,
                          const float* __restrict__ dinv,
                          const float* __restrict__ conv_b,
                          const float* __restrict__ out_w,
                          const float* __restrict__ out_b,
                          float* __restrict__ out) {
    const int wid = threadIdx.x >> 5;
    const int lane = threadIdx.x & 31;
    const int v = blockIdx.x * 8 + wid;
    if (v >= NN) return;
    float acc0 = 0.f, acc1 = 0.f;
    #pragma unroll
    for (int t = 0; t < 8; ++t) {
        int j = t * 32 + lane;            // 0..255
        int band = j >> 6;
        float di = dinv[band * NN + v];
        float h = bandout[v * BH + j] + di * di * xw[v * BH + j] + conv_b[j];
        h = fmaxf(h, 0.f);
        acc0 += h * out_w[j * 2 + 0];
        acc1 += h * out_w[j * 2 + 1];
    }
    #pragma unroll
    for (int off = 16; off > 0; off >>= 1) {
        acc0 += __shfl_xor(acc0, off, 32);
        acc1 += __shfl_xor(acc1, off, 32);
    }
    if (lane == 0) {
        float l0 = acc0 + out_b[0];
        float l1 = acc1 + out_b[1];
        float m = fmaxf(l0, l1);
        float lse = m + logf(expf(l0 - m) + expf(l1 - m));
        out[v * 2 + 0] = l0 - lse;
        out[v * 2 + 1] = l1 - lse;
    }
}

extern "C" void kernel_launch(void* const* d_in, const int* in_sizes, int n_in,
                              void* d_out, int out_size, void* d_ws, size_t ws_size,
                              hipStream_t stream) {
    const float* feature    = (const float*)d_in[0];
    const int*   edge_index = (const int*)  d_in[1];
    const float* edge_attr  = (const float*)d_in[2];
    const int*   sub_edges  = (const int*)  d_in[3];
    const float* te_w       = (const float*)d_in[4];
    const float* te_b       = (const float*)d_in[5];
    const float* ta_w       = (const float*)d_in[6];
    const float* ta_b       = (const float*)d_in[7];
    const float* conv_w     = (const float*)d_in[8];
    const float* conv_b     = (const float*)d_in[9];
    const float* out_w      = (const float*)d_in[10];
    const float* out_b      = (const float*)d_in[11];
    float* out = (float*)d_out;

    char* ws = (char*)d_ws;
    size_t off = 0;
    auto alloc = [&](size_t bytes) -> char* {
        char* p = ws + off;
        off = (off + bytes + 255) & ~(size_t)255;
        return p;
    };
    float*    t_agg    = (float*)   alloc((size_t)NN * HH * 4);   // 12.8 MB
    float*    dinv     = (float*)   alloc((size_t)BB * NN * 4);   //  0.8 MB
    _Float16* featH    = (_Float16*)alloc((size_t)NN * FH * 2);   // 19.2 MB
    _Float16* taw_pack = (_Float16*)alloc((size_t)4096 * 2);
    _Float16* w_pack   = (_Float16*)alloc((size_t)49152 * 2);
    float*    xw       = (float*)   alloc((size_t)NN * BH * 4);   // 51.2 MB
    float*    bandout  = (float*)   alloc((size_t)NN * BH * 4);   // 51.2 MB

    const int TB = 256;
    // init scratch
    fill_f32<<<(NN * HH + TB - 1) / TB, TB, 0, stream>>>(t_agg, 0.0f, NN * HH);
    fill_f32<<<(NN * BH + TB - 1) / TB, TB, 0, stream>>>(bandout, 0.0f, NN * BH);
    fill_f32<<<(BB * NN + TB - 1) / TB, TB, 0, stream>>>(dinv, 1.0f, BB * NN); // self-loop

    // stage 1: time encode + scatter, degrees, weight packing, feature conversion
    time_encode_scatter<<<(EE * 64) / TB, TB, 0, stream>>>(edge_attr, edge_index, te_w, te_b, t_agg);
    band_degree<<<(BB * EE + TB - 1) / TB, TB, 0, stream>>>(sub_edges, dinv);
    deg_to_dinv<<<(BB * NN + TB - 1) / TB, TB, 0, stream>>>(dinv);
    pack_taw<<<(4096 + TB - 1) / TB, TB, 0, stream>>>(ta_w, taw_pack);
    pack_convw<<<(49152 + TB - 1) / TB, TB, 0, stream>>>(conv_w, w_pack);
    feat_to_half<<<(NN * FF + TB - 1) / TB, TB, 0, stream>>>(feature, featH);

    // stage 2: agg = t_agg @ ta_w + b  (WMMA, 8 per wave)  -> featH[:,128:192]
    agg_gemm<<<NN / 16, 32, 0, stream>>>(t_agg, taw_pack, ta_b, featH);

    // stage 3: fused 4-band GEMM (WMMA, 24 per wave): xw = featH @ Wcat
    band_gemm<<<NN / 16, 128, 0, stream>>>(featH, w_pack, xw);

    // stage 4: per-band normalized scatter over subgraph edges (L2-resident atomics)
    band_scatter<<<(BB * EE * 64) / TB, TB, 0, stream>>>(sub_edges, dinv, xw, bandout);

    // stage 5: self loops + bias + relu + output linear + log_softmax
    final_out<<<NN / 8, TB, 0, stream>>>(bandout, xw, dinv, conv_b, out_w, out_b, out);
}
